// EnhancedCondConv2d_6012954215045
// MI455X (gfx1250) — matrix operand order repaired and checked by reading, not compiled, not run on hardware
//
#include <hip/hip_runtime.h>

typedef __attribute__((ext_vector_type(16))) _Float16 v16h;
typedef __attribute__((ext_vector_type(8)))  float    v8f;
typedef int v4i __attribute__((vector_size(16)));          // matches builtin's param type
typedef __attribute__((address_space(3))) v4i lds_v4i;

#if defined(__has_builtin)
#  if __has_builtin(__builtin_amdgcn_global_load_async_to_lds_b128)
#    define HAVE_ASYNC_LDS 1
#  endif
#endif
#ifndef HAVE_ASYNC_LDS
#  define HAVE_ASYNC_LDS 0
#endif

#if HAVE_ASYNC_LDS
#warning "CDNA5 path: global_load_async_to_lds_b128 ENABLED"
#else
#warning "CDNA5 path: async-to-LDS builtin NOT available, using sync copy"
#endif

__device__ __forceinline__ void async_wait_all() {
#if HAVE_ASYNC_LDS
#  if __has_builtin(__builtin_amdgcn_s_wait_asynccnt)
    __builtin_amdgcn_s_wait_asynccnt(0);
#  else
    asm volatile("s_wait_asynccnt 0" ::: "memory");
#  endif
#endif
}

constexpr int Bn  = 32;
constexpr int CI  = 128;
constexpr int CO  = 256;
constexpr int Hh  = 64;
constexpr int Ww  = 64;
constexpr int En  = 8;
constexpr int HW  = Hh * Ww;   // 4096
constexpr int NT  = 32;        // spatial tiles of 128 per sample

// ---------------- mean-pool x over H,W: one block per (b,c) plane ----------------
__global__ void pool_plane_kernel(const float* __restrict__ src, float* __restrict__ dst) {
    __shared__ float red[256];
    const float* p = src + (size_t)blockIdx.x * HW;
    float s = 0.f;
    for (int i = threadIdx.x; i < HW; i += 256) s += p[i];
    red[threadIdx.x] = s; __syncthreads();
    for (int off = 128; off > 0; off >>= 1) {
        if (threadIdx.x < off) red[threadIdx.x] += red[threadIdx.x + off];
        __syncthreads();
    }
    if (threadIdx.x == 0) dst[blockIdx.x] = red[0] * (1.0f / HW);
}

// ---------------- routing MLP + softmax: one block (128 thr) per sample ----------------
__global__ void route_kernel(const float* __restrict__ pooled,
                             const float* __restrict__ rw1, const float* __restrict__ rb1,
                             const float* __restrict__ rw2, const float* __restrict__ rb2,
                             const float* __restrict__ rw3, const float* __restrict__ rb3,
                             float* __restrict__ rweights) {
    __shared__ float red[128], sp[128], sv[128], hb[8], lg[8];
    const int b = blockIdx.x, t = threadIdx.x;
    sp[t] = pooled[b * CI + t];
    __syncthreads();
    for (int j = 0; j < 8; ++j) {
        red[t] = sp[t] * rw1[j * CI + t];
        __syncthreads();
        for (int off = 64; off > 0; off >>= 1) { if (t < off) red[t] += red[t + off]; __syncthreads(); }
        if (t == 0) hb[j] = fmaxf(red[0] + rb1[j], 0.f);
        __syncthreads();
    }
    float acc = rb2[t];
    #pragma unroll
    for (int j = 0; j < 8; ++j) acc += hb[j] * rw2[t * 8 + j];
    sv[t] = 1.f / (1.f + __expf(-acc));
    __syncthreads();
    for (int e = 0; e < 8; ++e) {
        red[t] = sv[t] * rw3[e * CI + t];
        __syncthreads();
        for (int off = 64; off > 0; off >>= 1) { if (t < off) red[t] += red[t + off]; __syncthreads(); }
        if (t == 0) lg[e] = red[0] + rb3[e];
        __syncthreads();
    }
    if (t == 0) {
        float mx = lg[0];
        for (int e = 1; e < 8; ++e) mx = fmaxf(mx, lg[e]);
        float ex[8], sum = 0.f;
        for (int e = 0; e < 8; ++e) { ex[e] = __expf(lg[e] - mx); sum += ex[e]; }
        for (int e = 0; e < 8; ++e) rweights[b * 8 + e] = ex[e] / sum;
    }
}

// ---------------- expert combine -> f16 weights, layout [b][kh*3+kw][co][ci] ----------------
__global__ void combine_kernel(const float* __restrict__ rweights,
                               const float* __restrict__ experts,
                               _Float16* __restrict__ wc) {
    const int b = blockIdx.x / 9, kk9 = blockIdx.x % 9;
    __shared__ float rw[8];
    if (threadIdx.x < 8) rw[threadIdx.x] = rweights[b * 8 + threadIdx.x];
    __syncthreads();
    for (int idx = threadIdx.x; idx < CO * CI; idx += 256) {  // idx = co*CI + ci
        float acc = 0.f;
        #pragma unroll
        for (int e = 0; e < 8; ++e)
            acc += rw[e] * experts[(size_t)(e * CO * CI + idx) * 9 + kk9];
        wc[(size_t)blockIdx.x * CO * CI + idx] = (_Float16)acc;
    }
}

// ---------------- implicit-GEMM conv via v_wmma_f32_16x16x32_f16 ----------------
// grid: Bn * (CO/128) * (HW/128) = 2048 blocks, 256 threads (8 waves)
// block tile 128(M=co) x 128(N=hw); wave tile 64x32 = 4x2 wmma tiles
// double-buffered LDS; A tile staged with async global->LDS; fused output pooling.
__global__ __launch_bounds__(256) void condconv_wmma_kernel(
        const float* __restrict__ x, const _Float16* __restrict__ wc,
        float* __restrict__ out, float* __restrict__ pp /* [b][co][nTile] */) {
    __shared__ __attribute__((aligned(16))) _Float16 ldsA[2][128 * 32]; // [m][k]
    __shared__ __attribute__((aligned(16))) _Float16 ldsB[2][128 * 32]; // [n][k]
    __shared__ float scratch[4][128];  // per-nGroup partial channel sums

    const int bid   = blockIdx.x;
    const int b     = bid >> 6;
    const int rem   = bid & 63;
    const int mTile = rem >> 5;          // 0..1 -> co block of 128
    const int nTile = rem & 31;          // 0..31 -> 128 spatial (2 rows of 64)
    const int co_base = mTile * 128;
    const int h0      = nTile * 2;

    const int tid  = threadIdx.x;
    const int lane = tid & 31, wave = tid >> 5;
    const int mWaveBase = (wave >> 2) * 64;
    const int nWaveBase = (wave & 3) * 32;
    const int hi = lane >> 4, lo = lane & 15;

    // B-staging invariants per thread
    const int n  = tid & 127;
    const int kb = (tid >> 7) * 16;

    const float*    xb  = x  + (size_t)b * CI * HW;
    const _Float16* wcB = wc + (size_t)b * 9 * CO * CI + (size_t)co_base * CI;

    v8f acc[4][2];
    v8f zero = {};
    #pragma unroll
    for (int mi = 0; mi < 4; ++mi)
        #pragma unroll
        for (int ni = 0; ni < 2; ++ni) acc[mi][ni] = zero;

    // ---- staging helpers: K-step s in [0,36), kk9 = s>>2, ci0 = (s&3)*32 ----
    auto issueA = [&](int s, int buf) {
        const int kk9 = s >> 2, ci0 = (s & 3) * 32;
        const _Float16* wA = wcB + (size_t)kk9 * CO * CI + ci0;
        const int c0 = tid, c1 = tid + 256;   // 512 16B chunks: m = c>>2, quarter = c&3
#if HAVE_ASYNC_LDS
        __builtin_amdgcn_global_load_async_to_lds_b128(
            (v4i*)(wA + (c0 >> 2) * CI + (c0 & 3) * 8),
            (lds_v4i*)&ldsA[buf][(c0 >> 2) * 32 + (c0 & 3) * 8], 0, 0);
        __builtin_amdgcn_global_load_async_to_lds_b128(
            (v4i*)(wA + (c1 >> 2) * CI + (c1 & 3) * 8),
            (lds_v4i*)&ldsA[buf][(c1 >> 2) * 32 + (c1 & 3) * 8], 0, 0);
#else
        uint4 va = *(const uint4*)(wA + (c0 >> 2) * CI + (c0 & 3) * 8);
        uint4 vb = *(const uint4*)(wA + (c1 >> 2) * CI + (c1 & 3) * 8);
        *(uint4*)&ldsA[buf][(c0 >> 2) * 32 + (c0 & 3) * 8] = va;
        *(uint4*)&ldsA[buf][(c1 >> 2) * 32 + (c1 & 3) * 8] = vb;
#endif
    };
    auto loadB = [&](int s, float* br) {
        const int kk9 = s >> 2, ci0 = (s & 3) * 32;
        const int kh  = (kk9 * 11) >> 5;      // kk9/3 for 0..8
        const int kw  = kk9 - kh * 3;
        const int h_im = h0 + (n >> 6) + kh - 1;
        const int w_im = (n & 63) + kw - 1;
        const bool valid = ((unsigned)h_im < (unsigned)Hh) && ((unsigned)w_im < (unsigned)Ww);
        const float* xrow = xb + h_im * Ww + w_im;
        #pragma unroll
        for (int j = 0; j < 16; ++j) {
            const int ci = ci0 + kb + j;
            br[j] = valid ? xrow[(size_t)ci * HW] : 0.f;
        }
    };
    auto storeB = [&](const float* br, int buf) {
        union { _Float16 h[16]; uint4 u[2]; } tb;
        #pragma unroll
        for (int j = 0; j < 16; ++j) tb.h[j] = (_Float16)br[j];
        uint4* d = (uint4*)&ldsB[buf][n * 32 + kb];
        d[0] = tb.u[0]; d[1] = tb.u[1];
    };
    auto compute = [&](int buf) {
        v16h bfrag[2];
        #pragma unroll
        for (int ni = 0; ni < 2; ++ni) {
            const int nloc = nWaveBase + ni * 16 + lo;
            union { v16h v; uint4 u[2]; } ub;
            const uint4* p = (const uint4*)&ldsB[buf][nloc * 32 + hi * 16];
            ub.u[0] = p[0]; ub.u[1] = p[1];
            bfrag[ni] = ub.v;
        }
        #pragma unroll
        for (int mi = 0; mi < 4; ++mi) {
            const int mloc = mWaveBase + mi * 16 + lo;
            union { v16h v; uint4 u[2]; } ua;
            ua.u[0] = *(const uint4*)&ldsA[buf][mloc * 32 + hi * 16];
            ua.u[1] = *(const uint4*)&ldsA[buf][mloc * 32 + hi * 16 + 8];
            #pragma unroll
            for (int ni = 0; ni < 2; ++ni) {
                acc[mi][ni] = __builtin_amdgcn_wmma_f32_16x16x32_f16(
                    false, ua.v, false, bfrag[ni],
                    (short)0, acc[mi][ni], false, false);
            }
        }
    };

    // ---- software pipeline over 36 K-steps, double-buffered ----
    float br[16];
    issueA(0, 0); loadB(0, br); storeB(br, 0);
    async_wait_all();
    __syncthreads();
    for (int s = 0; s < 36; ++s) {
        const int cur = s & 1, nxt = cur ^ 1;
        const bool more = (s + 1) < 36;
        if (more) { issueA(s + 1, nxt); loadB(s + 1, br); }
        compute(cur);
        if (more) storeB(br, nxt);
        async_wait_all();
        __syncthreads();
    }

    // ---- epilogue: store D tiles (m = r + 8*hi, n = lane&15) ----
    const int nTileBase = nTile * 128;
    #pragma unroll
    for (int mi = 0; mi < 4; ++mi) {
        #pragma unroll
        for (int ni = 0; ni < 2; ++ni) {
            const int n_g = nTileBase + nWaveBase + ni * 16 + lo;
            const int h = n_g >> 6, w = n_g & 63;
            #pragma unroll
            for (int r = 0; r < 8; ++r) {
                const int m_g = co_base + mWaveBase + mi * 16 + hi * 8 + r;
                out[((size_t)(b * CO + m_g) * Hh + h) * Ww + w] = acc[mi][ni][r];
            }
        }
    }

    // ---- fused, deterministic channel pooling: wave reduce -> LDS -> pp slot ----
    #pragma unroll
    for (int mi = 0; mi < 4; ++mi) {
        #pragma unroll
        for (int r = 0; r < 8; ++r) {
            float v = acc[mi][0][r] + acc[mi][1][r];   // sum over this wave's 32 n
            v += __shfl_xor(v, 1, 32);
            v += __shfl_xor(v, 2, 32);
            v += __shfl_xor(v, 4, 32);
            v += __shfl_xor(v, 8, 32);                 // sum over lo within each 16-lane half
            if (lo == 0) {
                const int m_local = mWaveBase + mi * 16 + hi * 8 + r;
                scratch[wave & 3][m_local] = v;
            }
        }
    }
    __syncthreads();
    if (tid < 128) {
        const float s4 = scratch[0][tid] + scratch[1][tid] + scratch[2][tid] + scratch[3][tid];
        pp[(size_t)(b * CO + co_base + tid) * NT + nTile] = s4;
    }
}

// ---------------- channel-attention MLP: one block (256 thr) per sample ----------------
__global__ void attn_kernel(const float* __restrict__ pp,
                            const float* __restrict__ aw1, const float* __restrict__ ab1,
                            const float* __restrict__ aw2, const float* __restrict__ ab2,
                            float* __restrict__ ca) {
    __shared__ float red[256], sp[256], hb[16];
    const int b = blockIdx.x, t = threadIdx.x;
    float s = 0.f;
    #pragma unroll 4
    for (int nt = 0; nt < NT; ++nt) s += pp[(size_t)(b * CO + t) * NT + nt];
    sp[t] = s * (1.0f / HW);
    __syncthreads();
    for (int j = 0; j < 16; ++j) {
        red[t] = sp[t] * aw1[j * CO + t];
        __syncthreads();
        for (int off = 128; off > 0; off >>= 1) { if (t < off) red[t] += red[t + off]; __syncthreads(); }
        if (t == 0) hb[j] = fmaxf(red[0] + ab1[j], 0.f);
        __syncthreads();
    }
    float acc = ab2[t];
    #pragma unroll
    for (int j = 0; j < 16; ++j) acc += hb[j] * aw2[t * 16 + j];
    ca[b * CO + t] = 1.f / (1.f + __expf(-acc));
}

// ---------------- apply attention: out *= ca[b,co], float4 ----------------
__global__ void scale_kernel(float* __restrict__ out, const float* __restrict__ ca) {
    const size_t i4 = (size_t)blockIdx.x * 256 + threadIdx.x;  // float4 index
    const int plane = (int)(i4 >> 10);                         // 1024 float4 per HW plane
    const float s = ca[plane];
    float4* o = (float4*)out;
    float4 v = o[i4];
    v.x *= s; v.y *= s; v.z *= s; v.w *= s;
    o[i4] = v;
}

extern "C" void kernel_launch(void* const* d_in, const int* in_sizes, int n_in,
                              void* d_out, int out_size, void* d_ws, size_t ws_size,
                              hipStream_t stream) {
    const float* x       = (const float*)d_in[0];
    const float* experts = (const float*)d_in[1];
    const float* rw1 = (const float*)d_in[2];  const float* rb1 = (const float*)d_in[3];
    const float* rw2 = (const float*)d_in[4];  const float* rb2 = (const float*)d_in[5];
    const float* rw3 = (const float*)d_in[6];  const float* rb3 = (const float*)d_in[7];
    const float* aw1 = (const float*)d_in[8];  const float* ab1 = (const float*)d_in[9];
    const float* aw2 = (const float*)d_in[10]; const float* ab2 = (const float*)d_in[11];
    float* out = (float*)d_out;

    // workspace layout
    char* ws = (char*)d_ws;
    float*    pooled   = (float*)ws;                                   // Bn*CI
    float*    rweights = pooled + Bn * CI;                             // Bn*En
    _Float16* wcomb    = (_Float16*)(rweights + Bn * En);              // Bn*9*CO*CI f16
    float*    pp       = (float*)((char*)wcomb + (size_t)Bn * 9 * CO * CI * sizeof(_Float16)); // Bn*CO*NT
    float*    ca       = pp + (size_t)Bn * CO * NT;                    // Bn*CO

    pool_plane_kernel<<<Bn * CI, 256, 0, stream>>>(x, pooled);
    route_kernel<<<Bn, 128, 0, stream>>>(pooled, rw1, rb1, rw2, rb2, rw3, rb3, rweights);
    combine_kernel<<<Bn * 9, 256, 0, stream>>>(rweights, experts, wcomb);
    condconv_wmma_kernel<<<Bn * (CO / 128) * (HW / 128), 256, 0, stream>>>(x, wcomb, out, pp);
    attn_kernel<<<Bn, 256, 0, stream>>>(pp, aw1, ab1, aw2, ab2, ca);
    scale_kernel<<<(Bn * CO * HW) / (256 * 4), 256, 0, stream>>>(out, ca);
}
